// MultiHeadAttentionBlock_21629455303097
// MI455X (gfx1250) — matrix operand order repaired and checked by reading, not compiled, not run on hardware
//
#include <hip/hip_runtime.h>

// Problem dims (fixed by reference)
#define B_   8
#define SQ_  1024
#define SK_  1024
#define D_   1024
#define ND_  1024
#define H_   8
#define DS_  128   // head dim

typedef __attribute__((ext_vector_type(16))) __bf16 v16bf;
typedef __attribute__((ext_vector_type(8)))  __bf16 v8bf;
typedef __attribute__((ext_vector_type(8)))  float  v8f;

__device__ __forceinline__ unsigned short f2bf(float f) {
  unsigned int u = __float_as_uint(f);
  u += 0x7FFFu + ((u >> 16) & 1u);          // round-to-nearest-even
  return (unsigned short)(u >> 16);
}
__device__ __forceinline__ float bf2f(unsigned short s) {
  return __uint_as_float(((unsigned int)s) << 16);
}
__device__ __forceinline__ v16bf cat8(v8bf lo, v8bf hi) {
  return __builtin_shufflevector(lo, hi, 0,1,2,3,4,5,6,7,8,9,10,11,12,13,14,15);
}
__device__ __forceinline__ v8f v8f_zero() {
  v8f r;
#pragma unroll
  for (int i = 0; i < 8; ++i) r[i] = 0.f;
  return r;
}
__device__ __forceinline__ v8f wmma_bf16(v16bf a, v16bf b, v8f c) {
  // D = A(16x32 bf16) x B(32x16 bf16) + C(16x16 f32)
  return __builtin_amdgcn_wmma_f32_16x16x32_bf16(
      /*neg_a=*/false, a, /*neg_b=*/false, b,
      /*c_mod=*/(short)0, c, /*reuse_a=*/false, /*reuse_b=*/false);
}

// ---------------------------------------------------------------- f32 -> bf16
__global__ __launch_bounds__(256)
void cvt_kernel(const float* __restrict__ in, unsigned short* __restrict__ out,
                long long n) {
  long long i = (long long)blockIdx.x * blockDim.x + threadIdx.x;
  long long stride = (long long)gridDim.x * blockDim.x;
  for (; i < n; i += stride) out[i] = f2bf(in[i]);
}

// ------------------------------------------------ mask -> 1 bit per element
// out word w = ballot(mask[w*32 + lane] != 0)  (wave32)
__global__ __launch_bounds__(256)
void maskpack_kernel(const int* __restrict__ mask,
                     unsigned int* __restrict__ out, long long nwords) {
  const int lane = threadIdx.x & 31;
  long long w = ((long long)blockIdx.x * blockDim.x + threadIdx.x) >> 5;
  const long long wstride = ((long long)gridDim.x * blockDim.x) >> 5;
  for (; w < nwords; w += wstride) {
    unsigned long long bal = __ballot(mask[w * 32 + lane] != 0);
    if (lane == 0) out[w] = (unsigned int)bal;
  }
}

// ------------------------------------------------- C[M,N] = A[M,K] @ W[N,K]^T
// 8 waves: 4(M) x 2(N); wave tile 32x64 (2x4 WMMA accums); block tile 128x128.
// MODE 0: outb[m,n]  = bf16(acc + bias[n])
// MODE 1: outf[m,n]  = resid[m,n] + relu(acc + bias[n])
template <int MODE>
__global__ __launch_bounds__(256)
void gemm_kernel(const unsigned short* __restrict__ A,
                 const unsigned short* __restrict__ W,
                 const float* __restrict__ bias,
                 unsigned short* __restrict__ outb,
                 const float* __restrict__ resid,
                 float* __restrict__ outf,
                 int M, int N, int K) {
  const int lane  = threadIdx.x & 31;
  const int w     = threadIdx.x >> 5;
  const int mbase = blockIdx.x * 128 + (w & 3) * 32;
  const int nbase = blockIdx.y * 128 + (w >> 2) * 64;

  v8f acc[2][4];
#pragma unroll
  for (int mi = 0; mi < 2; ++mi)
#pragma unroll
    for (int ni = 0; ni < 4; ++ni) acc[mi][ni] = v8f_zero();

  const int arow = mbase + (lane & 15);
  const int brow = nbase + (lane & 15);
  const int aoff = (lane < 16) ? 0 : 16;          // A: 16 contiguous K per lane
  const int boff = (lane < 16) ? 0 : 16;          // B: 16 contiguous K per lane

  for (int k = 0; k < K; k += 32) {
    if (k + 64 < K) {                             // prefetch next-next K slab
      __builtin_prefetch(A + (size_t)arow * K + k + 64, 0, 1);
      __builtin_prefetch(W + (size_t)brow * K + k + 64, 0, 1);
    }
    v16bf af[2], bf[4];
#pragma unroll
    for (int mi = 0; mi < 2; ++mi) {
      const unsigned short* p = A + (size_t)(arow + 16 * mi) * K + k + aoff;
      af[mi] = *(const v16bf*)p;
    }
#pragma unroll
    for (int ni = 0; ni < 4; ++ni) {
      const unsigned short* p = W + (size_t)(brow + 16 * ni) * K + k + boff;
      bf[ni] = *(const v16bf*)p;
    }
#pragma unroll
    for (int mi = 0; mi < 2; ++mi)
#pragma unroll
      for (int ni = 0; ni < 4; ++ni)
        acc[mi][ni] = wmma_bf16(af[mi], bf[ni], acc[mi][ni]);
  }

  const int n0 = nbase + (lane & 15);
  const int m0 = mbase + 8 * (lane >> 4);
#pragma unroll
  for (int mi = 0; mi < 2; ++mi)
#pragma unroll
    for (int ni = 0; ni < 4; ++ni) {
      const int n = n0 + 16 * ni;
      const float bv = bias[n];
#pragma unroll
      for (int j = 0; j < 8; ++j) {
        const int m = m0 + 16 * mi + j;
        const size_t idx = (size_t)m * N + n;
        float v = acc[mi][ni][j] + bv;
        if (MODE == 0) outb[idx] = f2bf(v);
        else           outf[idx] = resid[idx] + fmaxf(v, 0.f);
      }
    }
}

// ------------------------------------------------------- attention per head
// block = 4 waves (64 queries) sharing K/V LDS tiles; grid (SQ/64, B, H)
__global__ __launch_bounds__(128)
void attn_kernel(const unsigned short* __restrict__ qb,
                 const unsigned short* __restrict__ kb,
                 const unsigned short* __restrict__ vb,
                 const unsigned int* __restrict__ mbits,
                 float* __restrict__ out) {
  const int qt   = blockIdx.x;
  const int b    = blockIdx.y;
  const int h    = blockIdx.z;
  const int lane = threadIdx.x & 31;
  const int w    = threadIdx.x >> 5;   // 0..3

  __shared__ __align__(32) unsigned short sK[32][128];   // keys row-major
  __shared__ __align__(32) unsigned short sVt[128][32];  // V transposed [feat][key]
  __shared__ __align__(32) unsigned short sP[4][16][32]; // per-wave P staging

  const int mbase = qt * 64 + w * 16;
  const int l15   = lane & 15;
  const int aoff  = (lane < 16) ? 0 : 16;
  const int boff  = (lane < 16) ? 0 : 16;

  // Q fragments for 16 queries x 128 features (4 K-steps of 32)
  v16bf qf[4];
  {
    const unsigned short* qrowp =
        qb + (size_t)(b * SQ_ + mbase + l15) * ND_ + h * DS_;
#pragma unroll
    for (int ks = 0; ks < 4; ++ks) {
      const unsigned short* p = qrowp + ks * 32 + aoff;
      qf[ks] = *(const v16bf*)p;
    }
  }

  v8f o[8];
#pragma unroll
  for (int i = 0; i < 8; ++i) o[i] = v8f_zero();
  float rm[8], rs[8];
#pragma unroll
  for (int j = 0; j < 8; ++j) { rm[j] = -3.0e38f; rs[j] = 0.f; }

  const float scale = 0.03125f;  // 1/sqrt(ND)
  const int qr0 = mbase + 8 * (lane >> 4);

  for (int kt = 0; kt < SK_ / 32; ++kt) {
    // cooperative staging of K tile and transposed V tile
    for (int i = threadIdx.x; i < 512; i += 128) {
      const int r = i >> 4;
      const int c = (i & 15) * 8;
      const size_t rowbase = (size_t)(b * SK_ + kt * 32 + r) * ND_ + h * DS_ + c;
      *(uint4*)&sK[r][c] = *(const uint4*)(kb + rowbase);
      union { uint4 u; unsigned short s[8]; } t;
      t.u = *(const uint4*)(vb + rowbase);
#pragma unroll
      for (int jj = 0; jj < 8; ++jj) sVt[c + jj][r] = t.s[jj];
    }
    __syncthreads();

    // S = Q K^T over 32 keys (two 16-key N-tiles)
    v8f s0 = v8f_zero(), s1 = v8f_zero();
#pragma unroll
    for (int ks = 0; ks < 4; ++ks) {
      v16bf bk0 = *(const v16bf*)&sK[l15][ks * 32 + boff];
      v16bf bk1 = *(const v16bf*)&sK[16 + l15][ks * 32 + boff];
      s0 = wmma_bf16(qf[ks], bk0, s0);
      s1 = wmma_bf16(qf[ks], bk1, s1);
    }

    // packed mask + online softmax (row = element j within lane half)
    float p0a[8], p1a[8], fold[8];
#pragma unroll
    for (int j = 0; j < 8; ++j) {
      const unsigned int mb =
          mbits[(size_t)(b * SQ_ + qr0 + j) * (SK_ / 32) + kt];
      float v0 = ((mb >> l15) & 1u)        ? s0[j] * scale : -1.0e9f;
      float v1 = ((mb >> (16 + l15)) & 1u) ? s1[j] * scale : -1.0e9f;
      float tmax = fmaxf(v0, v1);
#pragma unroll
      for (int off = 1; off < 16; off <<= 1)
        tmax = fmaxf(tmax, __shfl_xor(tmax, off, 32));
      const float rmn = fmaxf(rm[j], tmax);
      const float f   = __expf(rm[j] - rmn);
      rm[j] = rmn;
      const float p0 = __expf(v0 - rmn);
      const float p1 = __expf(v1 - rmn);
      float psum = p0 + p1;
#pragma unroll
      for (int off = 1; off < 16; off <<= 1)
        psum += __shfl_xor(psum, off, 32);
      rs[j] = rs[j] * f + psum;
      fold[j] = f; p0a[j] = p0; p1a[j] = p1;
    }

    // rescale running O
#pragma unroll
    for (int nf = 0; nf < 8; ++nf)
#pragma unroll
      for (int j = 0; j < 8; ++j) o[nf][j] *= fold[j];

    // stage P (C layout) to LDS, reread in A layout
#pragma unroll
    for (int j = 0; j < 8; ++j) {
      const int mj = 8 * (lane >> 4) + j;
      sP[w][mj][l15]      = f2bf(p0a[j]);
      sP[w][mj][16 + l15] = f2bf(p1a[j]);
    }
    const unsigned short* pp = &sP[w][l15][0];
    v16bf pa = *(const v16bf*)(pp + aoff);

    // O += P @ V
#pragma unroll
    for (int nf = 0; nf < 8; ++nf) {
      v16bf vf = *(const v16bf*)&sVt[nf * 16 + l15][boff];
      o[nf] = wmma_bf16(pa, vf, o[nf]);
    }
    __syncthreads();
  }

  // finalize: normalize + q residual, write [b, q, h*128+feat]
  float inv[8];
#pragma unroll
  for (int j = 0; j < 8; ++j) inv[j] = 1.f / rs[j];
#pragma unroll
  for (int nf = 0; nf < 8; ++nf) {
    const int feat = h * DS_ + nf * 16 + l15;
#pragma unroll
    for (int j = 0; j < 8; ++j) {
      const size_t idx = (size_t)(b * SQ_ + qr0 + j) * ND_ + feat;
      out[idx] = o[nf][j] * inv[j] + bf2f(qb[idx]);
    }
  }
}

// ---------------------------------------------------------------- layernorm
__global__ __launch_bounds__(256)
void ln_kernel(const float* __restrict__ in, const float* __restrict__ g,
               const float* __restrict__ be, float* __restrict__ outf,
               unsigned short* __restrict__ outb, int N) {
  const int row = blockIdx.x;
  const float* x = in + (size_t)row * N;
  float s = 0.f, ss = 0.f;
  for (int i = threadIdx.x; i < N; i += 256) {
    const float v = x[i]; s += v; ss += v * v;
  }
  __shared__ float r1[256], r2[256];
  r1[threadIdx.x] = s; r2[threadIdx.x] = ss;
  __syncthreads();
  for (int off = 128; off > 0; off >>= 1) {
    if (threadIdx.x < off) {
      r1[threadIdx.x] += r1[threadIdx.x + off];
      r2[threadIdx.x] += r2[threadIdx.x + off];
    }
    __syncthreads();
  }
  const float mu   = r1[0] / N;
  const float var  = r2[0] / N - mu * mu;
  const float rinv = rsqrtf(var + 1e-5f);
  for (int i = threadIdx.x; i < N; i += 256) {
    const float y = (x[i] - mu) * rinv * g[i] + be[i];
    outf[(size_t)row * N + i] = y;
    if (outb) outb[(size_t)row * N + i] = f2bf(y);
  }
}

// ------------------------------------------------------------------- driver
extern "C" void kernel_launch(void* const* d_in, const int* in_sizes, int n_in,
                              void* d_out, int out_size, void* d_ws,
                              size_t ws_size, hipStream_t stream) {
  const float* Q    = (const float*)d_in[0];
  const float* K    = (const float*)d_in[1];
  const int*   mask = (const int*)d_in[2];
  const float* Wq   = (const float*)d_in[3];
  const float* bq   = (const float*)d_in[4];
  const float* Wk   = (const float*)d_in[5];
  const float* bk   = (const float*)d_in[6];
  const float* Wv   = (const float*)d_in[7];
  const float* bv   = (const float*)d_in[8];
  const float* Wo   = (const float*)d_in[9];
  const float* bo   = (const float*)d_in[10];
  const float* g0   = (const float*)d_in[11];
  const float* b0   = (const float*)d_in[12];
  const float* g1   = (const float*)d_in[13];
  const float* b1   = (const float*)d_in[14];
  float* out = (float*)d_out;

  char* ws = (char*)d_ws;
  size_t off = 0;
  auto alloc = [&](size_t bytes) -> char* {
    char* p = ws + off;
    off += (bytes + 255) & ~(size_t)255;
    return p;
  };
  const size_t MQ = (size_t)B_ * SQ_;  // 8192 rows

  unsigned short* wq_b = (unsigned short*)alloc((size_t)ND_ * D_ * 2);
  unsigned short* wk_b = (unsigned short*)alloc((size_t)ND_ * D_ * 2);
  unsigned short* wv_b = (unsigned short*)alloc((size_t)ND_ * D_ * 2);
  unsigned short* wo_b = (unsigned short*)alloc((size_t)ND_ * ND_ * 2);
  unsigned short* Qb   = (unsigned short*)alloc(MQ * D_ * 2);
  unsigned short* Kb   = (unsigned short*)alloc(MQ * D_ * 2);
  unsigned short* qb   = (unsigned short*)alloc(MQ * ND_ * 2);
  unsigned short* kb   = (unsigned short*)alloc(MQ * ND_ * 2);
  unsigned short* vb   = (unsigned short*)alloc(MQ * ND_ * 2);
  float* ln0f          = (float*)alloc(MQ * ND_ * 4);
  unsigned short* ln0b = (unsigned short*)alloc(MQ * ND_ * 2);
  unsigned int* mbits  = (unsigned int*)alloc(MQ * (SK_ / 32) * 4);
  // attention f32 output aliases the (dead-by-then) Qb+Kb staging (32 MB)
  float* attnf = (float*)Qb;

  // 1. convert inputs/weights to bf16; pack mask to bits
  cvt_kernel<<<2048, 256, 0, stream>>>(Q,  Qb,   (long long)MQ * D_);
  cvt_kernel<<<2048, 256, 0, stream>>>(K,  Kb,   (long long)MQ * D_);
  cvt_kernel<<<1024, 256, 0, stream>>>(Wq, wq_b, (long long)ND_ * D_);
  cvt_kernel<<<1024, 256, 0, stream>>>(Wk, wk_b, (long long)ND_ * D_);
  cvt_kernel<<<1024, 256, 0, stream>>>(Wv, wv_b, (long long)ND_ * D_);
  cvt_kernel<<<1024, 256, 0, stream>>>(Wo, wo_b, (long long)ND_ * ND_);
  maskpack_kernel<<<2048, 256, 0, stream>>>(mask, mbits,
                                            (long long)MQ * (SK_ / 32));

  // 2. q/k/v projections (bf16 out)
  dim3 gg((unsigned)(MQ / 128), ND_ / 128);
  gemm_kernel<0><<<gg, 256, 0, stream>>>(Qb, wq_b, bq, qb, nullptr, nullptr,
                                         (int)MQ, ND_, D_);
  gemm_kernel<0><<<gg, 256, 0, stream>>>(Kb, wk_b, bk, kb, nullptr, nullptr,
                                         (int)MQ, ND_, D_);
  gemm_kernel<0><<<gg, 256, 0, stream>>>(Kb, wv_b, bv, vb, nullptr, nullptr,
                                         (int)MQ, ND_, D_);

  // 3. attention (+ q residual) -> f32
  dim3 ga(SQ_ / 64, B_, H_);
  attn_kernel<<<ga, 128, 0, stream>>>(qb, kb, vb, mbits, attnf);

  // 4. LN0 -> f32 (residual) + bf16 (GEMM input)
  ln_kernel<<<dim3((unsigned)MQ), 256, 0, stream>>>(attnf, g0, b0, ln0f, ln0b, ND_);

  // 5. out = ln0 + relu(ln0 @ Wo^T + bo)  -> d_out
  gemm_kernel<1><<<gg, 256, 0, stream>>>(ln0b, wo_b, bo, nullptr, ln0f, out,
                                         (int)MQ, ND_, ND_);

  // 6. LN1 in place on d_out
  ln_kernel<<<dim3((unsigned)MQ), 256, 0, stream>>>(out, g1, b1, out, nullptr, ND_);
}
